// MMILossLayer_8383776162574
// MI455X (gfx1250) — matrix operand order repaired and checked by reading, not compiled, not run on hardware
//
#include <hip/hip_runtime.h>

// ---------------------------------------------------------------------------
// MMI loss (negative sum of pointwise MI estimate) via KDE self log-probs.
// N=8192, D=128.  d2_joint = d2_x + d2_y, so only two Gram matrices needed.
// Fused flash-style kernel: Gram tiles via v_wmma_f32_16x16x32_f16 with an
// f16 hi/lo split (~fp32 accuracy), streamed logsumexp over column tiles.
// Key simplification: every log-kernel term is c*d2 with c<0, d2>=0, and the
// diagonal term (j==i) gives d2=0 -> t=0, so the LSE max is exactly 0 a
// priori; no online max tracking needed, just sum of exp2(t) with t<=0.
// ---------------------------------------------------------------------------

typedef __attribute__((ext_vector_type(16))) _Float16 v16h;
typedef __attribute__((ext_vector_type(8)))  float    v8f;

#define NROWS   8192
#define DDIM    128
#define JCHUNKS 8
#define TILES_PER_CHUNK (NROWS / 16 / JCHUNKS)   // 64 column tiles per chunk

// fast transcendental builtins (v_exp_f32 / v_log_f32, both base-2)
#define EXP2F(x) __builtin_amdgcn_exp2f(x)
#define LOG2F(x) __builtin_amdgcn_logf(x)

__device__ __forceinline__ unsigned pack2h(_Float16 a, _Float16 b) {
    union { _Float16 h[2]; unsigned u; } u;
    u.h[0] = a; u.h[1] = b;
    return u.u;
}

// -------------------- kernel A: row squared norms ---------------------------
__global__ __launch_bounds__(256) void sqnorm_kernel(
    const float* __restrict__ X, const float* __restrict__ Y,
    float* __restrict__ sqx, float* __restrict__ sqy)
{
    int i = blockIdx.x * 256 + threadIdx.x;
    if (i >= NROWS) return;
    const float4* xr = (const float4*)(X + (size_t)i * DDIM);
    const float4* yr = (const float4*)(Y + (size_t)i * DDIM);
    float ax = 0.f, ay = 0.f;
    #pragma unroll 8
    for (int k = 0; k < DDIM / 4; ++k) {
        float4 vx = xr[k], vy = yr[k];
        ax += vx.x * vx.x + vx.y * vx.y + vx.z * vx.z + vx.w * vx.w;
        ay += vy.x * vy.x + vy.y * vy.y + vy.z * vy.z + vy.w * vy.w;
    }
    sqx[i] = ax;
    sqy[i] = ay;
}

// -------------------- kernel B: fused WMMA gram + exp2 sums -----------------
// grid: (64 row-blocks, JCHUNKS).  block: 256 threads = 8 waves.
// Wave w owns rows [rb*128 + w*16, +16).  All waves share staged column tiles.
__global__ __launch_bounds__(256) void kde_main_kernel(
    const float* __restrict__ X, const float* __restrict__ Y,
    const float* __restrict__ sjp, const float* __restrict__ sxp,
    const float* __restrict__ syp,
    const float* __restrict__ sqx, const float* __restrict__ sqy,
    float* __restrict__ partials)
{
    // 16 B-fragments (2 mats x hi/lo x 4 k-chunks), each 32 lanes x 16 halves
    __shared__ __align__(32) _Float16 sB[16 * 512];

    const int tid  = threadIdx.x;
    const int lane = tid & 31;
    const int wave = tid >> 5;
    const int rb   = blockIdx.x;           // 0..63
    const int jc   = blockIdx.y;           // 0..JCHUNKS-1
    const int i0   = rb * 128 + wave * 16; // wave's first row
    const int mrow = lane & 15;
    const int kh   = lane >> 4;

    const float LOG2E = 1.4426950408889634f;
    const float sx = sxp[0], sy = syp[0], sj = sjp[0];
    const float cxl = -0.5f * LOG2E / (sx * sx);
    const float cyl = -0.5f * LOG2E / (sy * sy);
    const float cjl = -0.5f * LOG2E / (sj * sj);

    // ---- build resident A fragments (16x32 f16 layout, ISA 7.12.2) --------
    // lane l: row M = l&15, k-half = l>>4 ; half h -> k = (h>=8?16:0)+kh*8+(h&7)
    v16h axh[4], axl[4], ayh[4], ayl[4];
    #pragma unroll
    for (int kc = 0; kc < 4; ++kc) {
        #pragma unroll
        for (int h = 0; h < 16; ++h) {
            int k = kc * 32 + ((h >= 8) ? 16 : 0) + kh * 8 + (h & 7);
            float vx = X[(size_t)(i0 + mrow) * DDIM + k];
            float vy = Y[(size_t)(i0 + mrow) * DDIM + k];
            _Float16 hx = (_Float16)vx, hy = (_Float16)vy;
            axh[kc][h] = hx; axl[kc][h] = (_Float16)(vx - (float)hx);
            ayh[kc][h] = hy; ayl[kc][h] = (_Float16)(vy - (float)hy);
        }
    }

    // squared norms of this lane's 8 rows (rows r + 8*kh)
    float sqxi[8], sqyi[8];
    #pragma unroll
    for (int r = 0; r < 8; ++r) {
        int i = i0 + r + 8 * kh;
        sqxi[r] = sqx[i];
        sqyi[r] = sqy[i];
    }

    // exp2-sum state (LSE max is identically 0): [0..7]=x, [8..15]=y, [16..23]=joint
    float sAcc[24];
    #pragma unroll
    for (int t = 0; t < 24; ++t) sAcc[t] = 0.f;

    const v16h* frag = (const v16h*)sB;   // frag f = sB + f*512 halves
    const int jt0 = jc * TILES_PER_CHUNK;

    for (int jt = jt0; jt < jt0 + TILES_PER_CHUNK; ++jt) {
        __syncthreads();   // previous tile fully consumed
        // ---- cooperative stage: convert tile to f16 hi/lo in B-frag layout
        // B layout: lane = n + 16*((kk>>3)&1), half h = (kk&7) + 8*(kk>>4)  (kk = k within chunk)
        #pragma unroll
        for (int p8 = 0; p8 < 8; ++p8) {
            int p   = p8 * 256 + tid;         // 0..2047 k-pairs
            int mat = p >> 10;                // 0=x, 1=y
            int jl  = (p >> 6) & 15;          // local column row
            int k   = (p & 63) * 2;           // even k
            const float* src = (mat == 0) ? X : Y;
            float2 v = *(const float2*)(src + (size_t)(jt * 16 + jl) * DDIM + k);
            _Float16 h0 = (_Float16)v.x, h1 = (_Float16)v.y;
            _Float16 l0 = (_Float16)(v.x - (float)h0);
            _Float16 l1 = (_Float16)(v.y - (float)h1);
            int kc = k >> 5, kk = k & 31;
            int bl = jl + 16 * ((kk >> 3) & 1);
            int hh = (kk & 7) + 8 * (kk >> 4); // even -> pair (hh, hh+1)
            int baseh = ((mat * 2 + 0) * 4 + kc) * 512 + bl * 16 + hh;
            *(unsigned*)&sB[baseh]            = pack2h(h0, h1);   // hi part
            *(unsigned*)&sB[baseh + 4 * 512]  = pack2h(l0, l1);   // lo part
        }
        __syncthreads();

        float sqxj = sqx[jt * 16 + mrow];
        float sqyj = sqy[jt * 16 + mrow];

        // ---- Gram tiles via WMMA with hi/lo split --------------------------
        v8f cx = {}, cy = {};
        #pragma unroll
        for (int kc = 0; kc < 4; ++kc) {
            v16h bh = frag[(0 * 4 + kc) * 32 + lane];   // x hi
            v16h bl = frag[(1 * 4 + kc) * 32 + lane];   // x lo
            cx = __builtin_amdgcn_wmma_f32_16x16x32_f16(false, axh[kc], false, bh,
                                                        (short)0, cx, false, false);
            cx = __builtin_amdgcn_wmma_f32_16x16x32_f16(false, axh[kc], false, bl,
                                                        (short)0, cx, false, false);
            cx = __builtin_amdgcn_wmma_f32_16x16x32_f16(false, axl[kc], false, bh,
                                                        (short)0, cx, false, false);
        }
        #pragma unroll
        for (int kc = 0; kc < 4; ++kc) {
            v16h bh = frag[(2 * 4 + kc) * 32 + lane];   // y hi
            v16h bl = frag[(3 * 4 + kc) * 32 + lane];   // y lo
            cy = __builtin_amdgcn_wmma_f32_16x16x32_f16(false, ayh[kc], false, bh,
                                                        (short)0, cy, false, false);
            cy = __builtin_amdgcn_wmma_f32_16x16x32_f16(false, ayh[kc], false, bl,
                                                        (short)0, cy, false, false);
            cy = __builtin_amdgcn_wmma_f32_16x16x32_f16(false, ayl[kc], false, bh,
                                                        (short)0, cy, false, false);
        }

        // ---- accumulate exp2 sums (max identically 0, terms <= 1) ----------
        #pragma unroll
        for (int r = 0; r < 8; ++r) {
            float d2x = fmaxf(sqxi[r] + sqxj - 2.f * cx[r], 0.f);
            float d2y = fmaxf(sqyi[r] + sqyj - 2.f * cy[r], 0.f);
            sAcc[r]      += EXP2F(cxl * d2x);
            sAcc[8 + r]  += EXP2F(cyl * d2y);
            sAcc[16 + r] += EXP2F(cjl * (d2x + d2y));
        }
    }

    // ---- sum partial sums across the 16 lanes sharing each row ------------
    #pragma unroll
    for (int t = 0; t < 24; ++t) {
        float sv = sAcc[t];
        #pragma unroll
        for (int msk = 1; msk < 16; msk <<= 1)
            sv += __shfl_xor(sv, msk, 32);
        sAcc[t] = sv;
    }

    if (mrow == 0) {
        #pragma unroll
        for (int r = 0; r < 8; ++r) {
            int i = i0 + r + 8 * kh;
            size_t base = ((size_t)jc * NROWS + i) * 3;
            partials[base + 0] = sAcc[r];
            partials[base + 1] = sAcc[8 + r];
            partials[base + 2] = sAcc[16 + r];
        }
    }
}

// -------------------- kernel C: merge chunks, per-row MI, block sums --------
__global__ __launch_bounds__(256) void finalize_kernel(
    const float* __restrict__ partials,
    const float* __restrict__ sjp, const float* __restrict__ sxp,
    const float* __restrict__ syp,
    float* __restrict__ blocksums)
{
    __shared__ float red[8];
    int i = blockIdx.x * 256 + threadIdx.x;   // row

    float s0 = 0.f, s1 = 0.f, s2 = 0.f;
    for (int jc = 0; jc < JCHUNKS; ++jc) {
        const float* p = partials + ((size_t)jc * NROWS + i) * 3;
        s0 += p[0];
        s1 += p[1];
        s2 += p[2];
    }

    const float LN2     = 0.6931471805599453f;
    const float LOG2E   = 1.4426950408889634f;
    const float LOG_2PI = 1.8378770664093453f;
    const float sx = sxp[0], sy = syp[0], sj = sjp[0];
    // lp = (log2 s - log2 N) * ln2 - d*ln(sigma) - d/2*ln(2pi);  N = 2^13
    float lnsx = LOG2F(sx) * LN2, lnsy = LOG2F(sy) * LN2, lnsj = LOG2F(sj) * LN2;
    float lpx = (LOG2F(s0) - 13.f) * LN2 - 128.f * lnsx -  64.f * LOG_2PI;
    float lpy = (LOG2F(s1) - 13.f) * LN2 - 128.f * lnsy -  64.f * LOG_2PI;
    float lpj = (LOG2F(s2) - 13.f) * LN2 - 256.f * lnsj - 128.f * LOG_2PI;
    float I = EXP2F(lpj * LOG2E) * (lpj - lpx - lpy);   // exp(lpj)*(...)

    // deterministic fixed-order reduction (no float atomics)
    float v = I;
    #pragma unroll
    for (int msk = 16; msk >= 1; msk >>= 1) v += __shfl_xor(v, msk, 32);
    if ((threadIdx.x & 31) == 0) red[threadIdx.x >> 5] = v;
    __syncthreads();
    if (threadIdx.x == 0) {
        float acc = 0.f;
        #pragma unroll
        for (int w = 0; w < 8; ++w) acc += red[w];
        blocksums[blockIdx.x] = acc;
    }
}

// -------------------- kernel D: final 32-way sum ----------------------------
__global__ void final_reduce_kernel(const float* __restrict__ blocksums,
                                    float* __restrict__ out)
{
    float v = blocksums[threadIdx.x];   // 32 threads
    #pragma unroll
    for (int msk = 16; msk >= 1; msk >>= 1) v += __shfl_xor(v, msk, 32);
    if (threadIdx.x == 0) out[0] = -v;  // loss = -sum(I_xy)
}

// ---------------------------------------------------------------------------
extern "C" void kernel_launch(void* const* d_in, const int* in_sizes, int n_in,
                              void* d_out, int out_size, void* d_ws, size_t ws_size,
                              hipStream_t stream)
{
    (void)in_sizes; (void)n_in; (void)out_size; (void)ws_size;
    const float* X  = (const float*)d_in[0];   // latent_x [8192,128]
    const float* Y  = (const float*)d_in[1];   // latent_y [8192,128]
    const float* sj = (const float*)d_in[2];   // sigma_joint
    const float* sx = (const float*)d_in[3];   // sigma_x
    const float* sy = (const float*)d_in[4];   // sigma_y
    float* out = (float*)d_out;

    float* sqx      = (float*)d_ws;
    float* sqy      = sqx + NROWS;
    float* partials = sqy + NROWS;                       // JCHUNKS*NROWS*3 floats
    float* bsums    = partials + (size_t)JCHUNKS * NROWS * 3; // 32 floats

    sqnorm_kernel<<<NROWS / 256, 256, 0, stream>>>(X, Y, sqx, sqy);

    dim3 grid(64, JCHUNKS);
    kde_main_kernel<<<grid, 256, 0, stream>>>(X, Y, sj, sx, sy, sqx, sqy, partials);

    finalize_kernel<<<NROWS / 256, 256, 0, stream>>>(partials, sj, sx, sy, bsums);
    final_reduce_kernel<<<1, 32, 0, stream>>>(bsums, out);
}
